// GLM45VBackbone_32813550141639
// MI455X (gfx1250) — matrix operand order repaired and checked by reading, not compiled, not run on hardware
//
#include <hip/hip_runtime.h>
#include <hip/hip_bf16.h>

// ---------------------------------------------------------------------------
// MoE backbone for MI455X (gfx1250, wave32, WMMA).
// - route top-2 experts (3.6x FLOP reduction vs dense)
// - one bandwidth-bound pre-pass: f32 -> bf16, weights transposed to [N][K]
// - all GEMMs: 128x128 block tiles, v_wmma_f32_16x16x32_bf16, f32 accumulate
// - tile staging: GLOBAL_LOAD_ASYNC_TO_LDS_B128 (ASYNCcnt) + double-buffered
//   LDS; branch-free unroll-by-2 pipeline with peeled tail.
// Workspace requirement: ~560 MB.
// ---------------------------------------------------------------------------

#define NTOK   8192
#define EMB    2048
#define HID    4096
#define NEXP   8
#define RMAX   (NTOK * 2 + 128)   // gathered rows (16384) + tile padding

typedef __attribute__((ext_vector_type(16))) __bf16 v16bf;
typedef __attribute__((ext_vector_type(8)))  __bf16 v8bf;
typedef __attribute__((ext_vector_type(8)))  float  v8f;

union ABFrag { v16bf v; v8bf h[2]; };

// ---------------------------------------------------------------------------
// small utility kernels
// ---------------------------------------------------------------------------
__global__ void init_counts(int* cnt, int* fill) {
    if (threadIdx.x < NEXP) { cnt[threadIdx.x] = 0; fill[threadIdx.x] = 0; }
}

__global__ void scan8(const int* __restrict__ cnt, int* __restrict__ offs) {
    if (threadIdx.x == 0) {
        int s = 0;
        for (int e = 0; e < NEXP; ++e) { offs[e] = s; s += cnt[e]; }
    }
}

// elementwise f32 -> bf16 (8 elems / thread)
__global__ __launch_bounds__(256)
void cvt_bf16(const float* __restrict__ src, __bf16* __restrict__ dst, size_t n) {
    size_t i = ((size_t)blockIdx.x * blockDim.x + threadIdx.x) * 8;
    if (i >= n) return;
    float4 a = *(const float4*)(src + i);
    float4 b = *(const float4*)(src + i + 4);
    v8bf o;
    o[0] = (__bf16)a.x; o[1] = (__bf16)a.y; o[2] = (__bf16)a.z; o[3] = (__bf16)a.w;
    o[4] = (__bf16)b.x; o[5] = (__bf16)b.y; o[6] = (__bf16)b.z; o[7] = (__bf16)b.w;
    *(v8bf*)(dst + i) = o;
}

// transpose + convert: src f32 [R][C] -> dst bf16 [C][R]   (R,C multiples of 64)
__global__ __launch_bounds__(256)
void tconv(const float* __restrict__ src, __bf16* __restrict__ dst,
           int R, int C, size_t sStride, size_t dStride) {
    src += (size_t)blockIdx.z * sStride;
    dst += (size_t)blockIdx.z * dStride;
    __shared__ float tile[64][65];
    int tr = blockIdx.y * 64;            // src row base
    int tc = blockIdx.x * 64;            // src col base
    int r    = threadIdx.x >> 2;         // 0..63
    int cblk = (threadIdx.x & 3) * 16;   // 0,16,32,48
    const float* sp = src + (size_t)(tr + r) * C + tc + cblk;
    #pragma unroll
    for (int i = 0; i < 16; i += 4) {
        float4 v = *(const float4*)(sp + i);
        tile[r][cblk + i + 0] = v.x; tile[r][cblk + i + 1] = v.y;
        tile[r][cblk + i + 2] = v.z; tile[r][cblk + i + 3] = v.w;
    }
    __syncthreads();
    // dst[tc+r][tr+cblk+i] = src[tr+cblk+i][tc+r] = tile[cblk+i][r]
    v8bf o0, o1;
    #pragma unroll
    for (int i = 0; i < 8; ++i) o0[i] = (__bf16)tile[cblk + i][r];
    #pragma unroll
    for (int i = 0; i < 8; ++i) o1[i] = (__bf16)tile[cblk + 8 + i][r];
    __bf16* dp = dst + (size_t)(tc + r) * R + tr + cblk;
    *(v8bf*)(dp)     = o0;
    *(v8bf*)(dp + 8) = o1;
}

// ---------------------------------------------------------------------------
// router: one wave32 per token; 8-way dot product + top-2 + softmax
// ---------------------------------------------------------------------------
__global__ __launch_bounds__(256)
void router(const float* __restrict__ h, const float* __restrict__ gw,
            const float* __restrict__ gb, int* __restrict__ tki,
            float* __restrict__ tkw, int* __restrict__ cnt) {
    int wv = threadIdx.x >> 5, ln = threadIdx.x & 31;
    int n = blockIdx.x * 8 + wv;
    const float* hr = h + (size_t)n * EMB;
    float acc[NEXP];
    #pragma unroll
    for (int e = 0; e < NEXP; ++e) acc[e] = 0.f;
    for (int k = ln; k < EMB; k += 32) {
        float x = hr[k];
        const float* g = gw + (size_t)k * NEXP;
        #pragma unroll
        for (int e = 0; e < NEXP; ++e) acc[e] += x * g[e];
    }
    #pragma unroll
    for (int off = 16; off > 0; off >>= 1) {
        #pragma unroll
        for (int e = 0; e < NEXP; ++e) acc[e] += __shfl_xor(acc[e], off, 32);
    }
    if (ln == 0) {
        float best = -1e30f, sec = -1e30f; int bi = 0, si = 1;
        #pragma unroll
        for (int e = 0; e < NEXP; ++e) {
            float v = acc[e] + gb[e];
            if (v > best)      { sec = best; si = bi; best = v; bi = e; }
            else if (v > sec)  { sec = v; si = e; }
        }
        float ex = __expf(sec - best);           // <= 1
        float w0 = 1.f / (1.f + ex);
        float w1 = ex * w0;
        tki[2 * n] = bi; tki[2 * n + 1] = si;
        tkw[2 * n] = w0; tkw[2 * n + 1] = w1;
        atomicAdd(&cnt[bi], 1);
        atomicAdd(&cnt[si], 1);
    }
}

__global__ __launch_bounds__(256)
void build_lists(const int* __restrict__ tki, const float* __restrict__ tkw,
                 const int* __restrict__ offs, int* __restrict__ fill,
                 int* __restrict__ list, float* __restrict__ wlist,
                 int* __restrict__ rowof) {
    int n = blockIdx.x * 256 + threadIdx.x;
    if (n >= NTOK) return;
    #pragma unroll
    for (int k = 0; k < 2; ++k) {
        int e = tki[2 * n + k];
        int p = atomicAdd(&fill[e], 1);
        int r = offs[e] + p;
        list[r]  = n;
        wlist[r] = tkw[2 * n + k];
        rowof[2 * n + k] = r;
    }
}

// ---------------------------------------------------------------------------
// WMMA GEMM: C[M x N] = A[M x K] * B^T-stored[N x K], bf16 in / f32 acc.
// MODE 0: gathered tokens -> hid   (epilogue: +b1, exact GELU, bf16 store)
// MODE 1: hid -> y                 (epilogue: +b2, * router weight, bf16 store)
// MODE 2: moe -> d_out             (epilogue: +proj_b, f32 store)
// Block tile 128x128, k-step 32, 8 waves of 32x64 (8 wmma each / k-step).
// Staging: async global->LDS DMA (b128), double-buffered, ASYNCcnt-tracked,
// unroll-by-2 branch-free steady state + peeled tail (requires K % 64 == 0).
// ---------------------------------------------------------------------------
template <int MODE>
__global__ __launch_bounds__(256)
void gemm_wmma(const __bf16* __restrict__ A, const __bf16* __restrict__ B,
               const float* __restrict__ bias, float* __restrict__ outF,
               __bf16* __restrict__ outH, const int* __restrict__ list,
               const float* __restrict__ wlist, const int* __restrict__ cnt,
               const int* __restrict__ offs, int K, int N) {
    __shared__ __align__(16) __bf16 Alds[2][128 * 40];   // 32 k + 8 pad / row
    __shared__ __align__(16) __bf16 Blds[2][128 * 40];
    __shared__ int tokLds[128];

    int rowStart, rem;
    if (MODE == 2) {
        rowStart = blockIdx.x * 128;
        rem = 128;
    } else {
        int e  = blockIdx.x >> 6;
        int lt = blockIdx.x & 63;
        int base = lt * 128;
        int c = cnt[e];
        if (base >= c) return;
        rem = c - base; if (rem > 128) rem = 128;
        rowStart = offs[e] + base;
        B    += (size_t)e * N * K;
        bias += (size_t)e * N;
    }
    int colStart = blockIdx.y * 128;

    int t = threadIdx.x;
    if (MODE == 0) {
        if (t < 128) tokLds[t] = (t < rem) ? list[rowStart + t] : list[rowStart];
        __syncthreads();
    }

    // ---- loop-invariant staging addresses (2 threads per tile row) ----
    int srow = t >> 1, shalf = t & 1;
    size_t arowBase = (MODE == 0) ? (size_t)tokLds[srow] * K
                                  : (size_t)(rowStart + srow) * K;
    const __bf16* aptr = A + arowBase + shalf * 16;
    const __bf16* bptr = B + (size_t)(colStart + srow) * K + shalf * 16;
    uint32_t soff = (uint32_t)(srow * 40 + shalf * 16);
    uint32_t ldsA[2] = {(uint32_t)(uintptr_t)&Alds[0][soff],
                        (uint32_t)(uintptr_t)&Alds[1][soff]};
    uint32_t ldsB[2] = {(uint32_t)(uintptr_t)&Blds[0][soff],
                        (uint32_t)(uintptr_t)&Blds[1][soff]};

    // one k-step of async DMA into LDS buffer b (4 x b128 per thread;
    // inst offset applies to both the LDS and global addresses)
    auto stage = [&](int b, int kk) {
        uint64_t ga = (uint64_t)(uintptr_t)(aptr + kk);
        uint64_t gb = (uint64_t)(uintptr_t)(bptr + kk);
        asm volatile(
            "s_clause 0x3\n\t"
            "global_load_async_to_lds_b128 %0, %2, off\n\t"
            "global_load_async_to_lds_b128 %0, %2, off offset:16\n\t"
            "global_load_async_to_lds_b128 %1, %3, off\n\t"
            "global_load_async_to_lds_b128 %1, %3, off offset:16"
            :: "v"(ldsA[b]), "v"(ldsB[b]), "v"(ga), "v"(gb)
            : "memory");
    };

    int wv = t >> 5, ln = t & 31;
    int wm = (wv >> 1) * 32;      // wave row offset in tile (0,32,64,96)
    int wn = (wv & 1) * 64;       // wave col offset in tile (0,64)
    int l15 = ln & 15, lh = ln >> 4;

    v8f acc[2][4];
    #pragma unroll
    for (int mt = 0; mt < 2; ++mt)
        #pragma unroll
        for (int nt = 0; nt < 4; ++nt)
            acc[mt][nt] = (v8f){0.f, 0.f, 0.f, 0.f, 0.f, 0.f, 0.f, 0.f};

    // fragment load + 8 WMMA from LDS buffer b (ISA 16-bit A/B layouts)
    auto computeStep = [&](int b) {
        ABFrag af[2], bfr[4];
        #pragma unroll
        for (int mt = 0; mt < 2; ++mt) {
            const __bf16* p = &Alds[b][(wm + mt * 16 + l15) * 40 + lh * 8];
            af[mt].h[0] = *(const v8bf*)(p);        // K 0-7  / 8-15
            af[mt].h[1] = *(const v8bf*)(p + 16);   // K 16-23/ 24-31
        }
        #pragma unroll
        for (int nt = 0; nt < 4; ++nt) {
            const __bf16* p = &Blds[b][(wn + nt * 16 + l15) * 40 + lh * 8];
            bfr[nt].h[0] = *(const v8bf*)(p);       // K 0-7  / 8-15
            bfr[nt].h[1] = *(const v8bf*)(p + 16);  // K 16-23/ 24-31
        }
        #pragma unroll
        for (int mt = 0; mt < 2; ++mt)
            #pragma unroll
            for (int nt = 0; nt < 4; ++nt)
                acc[mt][nt] = __builtin_amdgcn_wmma_f32_16x16x32_bf16(
                    false, af[mt].v, false, bfr[nt].v,
                    (short)0, acc[mt][nt], false, false);
    };

    // ---- software pipeline: 8 async b128 in flight, retire 4 per step ----
    stage(0, 0);
    stage(1, 32);
    int k0 = 0;
    for (; k0 + 64 < K; k0 += 64) {
        asm volatile("s_wait_asynccnt 0x4" ::: "memory");  // buf0 ready
        __syncthreads();
        computeStep(0);
        __syncthreads();
        stage(0, k0 + 64);
        asm volatile("s_wait_asynccnt 0x4" ::: "memory");  // buf1 ready
        __syncthreads();
        computeStep(1);
        __syncthreads();
        stage(1, k0 + 96);
    }
    // tail: k0 == K - 64
    asm volatile("s_wait_asynccnt 0x4" ::: "memory");
    __syncthreads();
    computeStep(0);
    __syncthreads();
    asm volatile("s_wait_asynccnt 0x0" ::: "memory");
    __syncthreads();
    computeStep(1);

    // ---- epilogue ----
    #pragma unroll
    for (int mt = 0; mt < 2; ++mt) {
        #pragma unroll
        for (int j = 0; j < 8; ++j) {
            int rowL = wm + mt * 16 + lh * 8 + j;
            size_t gRow = (size_t)rowStart + rowL;
            float wrow = (MODE == 1) ? wlist[gRow] : 1.f;
            #pragma unroll
            for (int nt = 0; nt < 4; ++nt) {
                int gCol = colStart + wn + nt * 16 + l15;
                float v = acc[mt][nt][j] + bias[gCol];
                if (MODE == 0) {
                    v = 0.5f * v * (1.f + erff(v * 0.70710678118f));   // exact GELU
                    if (rowL < rem) outH[gRow * (size_t)N + gCol] = (__bf16)v;
                } else if (MODE == 1) {
                    v *= wrow;
                    if (rowL < rem) outH[gRow * (size_t)N + gCol] = (__bf16)v;
                } else {
                    outF[gRow * (size_t)N + gCol] = v;
                }
            }
        }
    }
}

// combine the two expert rows of each token -> bf16 moe activations
__global__ __launch_bounds__(256)
void combine(const __bf16* __restrict__ y, const int* __restrict__ rowof,
             __bf16* __restrict__ moe) {
    int n = blockIdx.x;
    size_t r0 = (size_t)rowof[2 * n];
    size_t r1 = (size_t)rowof[2 * n + 1];
    for (int c = threadIdx.x; c < EMB; c += 256) {
        float v = (float)y[r0 * EMB + c] + (float)y[r1 * EMB + c];
        moe[(size_t)n * EMB + c] = (__bf16)v;
    }
}

// ---------------------------------------------------------------------------
extern "C" void kernel_launch(void* const* d_in, const int* in_sizes, int n_in,
                              void* d_out, int out_size, void* d_ws, size_t ws_size,
                              hipStream_t stream) {
    const float* h_c    = (const float*)d_in[0];
    const float* gate_w = (const float*)d_in[1];
    const float* gate_b = (const float*)d_in[2];
    const float* w1     = (const float*)d_in[3];
    const float* b1     = (const float*)d_in[4];
    const float* w2     = (const float*)d_in[5];
    const float* b2     = (const float*)d_in[6];
    const float* proj_w = (const float*)d_in[7];
    const float* proj_b = (const float*)d_in[8];
    float* out = (float*)d_out;

    // ---- workspace carve (~560 MB total) ----
    char* p = (char*)d_ws;
    auto carve = [&](size_t bytes) {
        void* r = (void*)p;
        p += (bytes + 255) & ~(size_t)255;
        return r;
    };
    __bf16* hB   = (__bf16*)carve((size_t)NTOK * EMB * 2);
    __bf16* w1T  = (__bf16*)carve((size_t)NEXP * HID * EMB * 2);   // [e][4096][2048]
    __bf16* w2T  = (__bf16*)carve((size_t)NEXP * EMB * HID * 2);   // [e][2048][4096]
    __bf16* pjT  = (__bf16*)carve((size_t)EMB * EMB * 2);          // [out][in]
    __bf16* hid  = (__bf16*)carve((size_t)RMAX * HID * 2);
    __bf16* ybuf = (__bf16*)carve((size_t)RMAX * EMB * 2);
    __bf16* moe  = (__bf16*)carve((size_t)NTOK * EMB * 2);
    int*    tki  = (int*)carve((size_t)NTOK * 2 * 4);
    float*  tkw  = (float*)carve((size_t)NTOK * 2 * 4);
    int*    list = (int*)carve((size_t)RMAX * 4);
    float*  wls  = (float*)carve((size_t)RMAX * 4);
    int*    rowof= (int*)carve((size_t)NTOK * 2 * 4);
    int*    cnt  = (int*)carve(64);
    int*    offs = (int*)carve(64);
    int*    fill = (int*)carve(64);

    // 1) init + precision conversion / weight transposes (bandwidth-bound)
    init_counts<<<1, 32, 0, stream>>>(cnt, fill);
    cvt_bf16<<<(NTOK * EMB) / 2048, 256, 0, stream>>>(h_c, hB, (size_t)NTOK * EMB);
    tconv<<<dim3(HID / 64, EMB / 64, NEXP), 256, 0, stream>>>(
        w1, w1T, EMB, HID, (size_t)EMB * HID, (size_t)HID * EMB);
    tconv<<<dim3(EMB / 64, HID / 64, NEXP), 256, 0, stream>>>(
        w2, w2T, HID, EMB, (size_t)HID * EMB, (size_t)EMB * HID);
    tconv<<<dim3(EMB / 64, EMB / 64, 1), 256, 0, stream>>>(
        proj_w, pjT, EMB, EMB, (size_t)EMB * EMB, (size_t)EMB * EMB);

    // 2) routing
    router<<<NTOK / 8, 256, 0, stream>>>(h_c, gate_w, gate_b, tki, tkw, cnt);
    scan8<<<1, 32, 0, stream>>>(cnt, offs);
    build_lists<<<NTOK / 256, 256, 0, stream>>>(tki, tkw, offs, fill, list, wls, rowof);

    // 3) expert GEMMs on gathered rows (bf16 WMMA, f32 accumulate)
    gemm_wmma<0><<<dim3(NEXP * 64, HID / 128), 256, 0, stream>>>(
        hB, w1T, b1, nullptr, hid, list, nullptr, cnt, offs, EMB, HID);
    gemm_wmma<1><<<dim3(NEXP * 64, EMB / 128), 256, 0, stream>>>(
        hid, w2T, b2, nullptr, ybuf, nullptr, wls, cnt, offs, HID, EMB);

    // 4) weighted combine -> bf16 activations
    combine<<<NTOK, 256, 0, stream>>>(ybuf, rowof, moe);

    // 5) output projection -> f32
    gemm_wmma<2><<<dim3(NTOK / 128, EMB / 128), 256, 0, stream>>>(
        moe, pjT, proj_b, out, nullptr, nullptr, nullptr, nullptr, nullptr,
        EMB, EMB);
}
